// Transformer_87428354277405
// MI455X (gfx1250) — matrix operand order repaired and checked by reading, not compiled, not run on hardware
//
#include <hip/hip_runtime.h>

typedef __bf16 bhalf;
typedef __attribute__((ext_vector_type(16))) __bf16 v16bf;
typedef __attribute__((ext_vector_type(8)))  float  v8f;
typedef __attribute__((ext_vector_type(4)))  int    v4i;

#define BB   4
#define SS   2048
#define DD   1024
#define FFD  4096
#define HHN  16
#define DHH  64
#define LLN  6
#define MTOK (BB*SS)

constexpr int F_BIAS  = 1;
constexpr int F_RESID = 2;
constexpr int F_RELU  = 4;
constexpr int F_BF16  = 8;

// ---------------------------------------------------------------------------
// CDNA5 async global->LDS copy (ASYNCcnt-tracked). Falls back to a
// synchronous float4 copy if the toolchain lacks the builtin.
// ---------------------------------------------------------------------------
#if defined(__HIP_DEVICE_COMPILE__) && \
    __has_builtin(__builtin_amdgcn_global_load_async_to_lds_b128)
#define HAVE_ASYNC 1
#else
#define HAVE_ASYNC 0
#endif

#define AS1 __attribute__((address_space(1)))
#define AS3 __attribute__((address_space(3)))

__device__ __forceinline__ void copy16(const bhalf* g, bhalf* l){
#if HAVE_ASYNC
  __builtin_amdgcn_global_load_async_to_lds_b128((AS1 v4i*)g, (AS3 v4i*)l, 0, 0);
#else
  *(float4*)l = *(const float4*)g;
#endif
}
__device__ __forceinline__ void async_wait0(){
#if HAVE_ASYNC
#if __has_builtin(__builtin_amdgcn_s_wait_asynccnt)
  __builtin_amdgcn_s_wait_asynccnt(0);
#else
  asm volatile("s_wait_asynccnt 0x0" ::: "memory");
#endif
#endif
}

// ---------------------------------------------------------------------------
// WMMA helpers (CDNA5 wave32, 16x16x32 bf16 -> f32)
// ---------------------------------------------------------------------------
__device__ __forceinline__ v8f wmma_bf16(v16bf a, v16bf b, v8f c){
  return __builtin_amdgcn_wmma_f32_16x16x32_bf16(false, a, false, b, (short)0, c,
                                                 false, false);
}

// A-fragment: 16x32 tile, row-major, stride ld (elements).
__device__ __forceinline__ v16bf frag_a(const bhalf* t, int ld){
  const int l = threadIdx.x & 31, r = l & 15, h = l >> 4;
  v16bf f;
  ((float4*)&f)[0] = *(const float4*)(t + r*ld + h*8);
  ((float4*)&f)[1] = *(const float4*)(t + r*ld + 16 + h*8);
  return f;
}
// B-fragment from a TRANSPOSED tile: Bt[n][k] row-major, stride ld.
__device__ __forceinline__ v16bf frag_b(const bhalf* t, int ld){
  const int l = threadIdx.x & 31, n = l & 15, h = l >> 4;
  v16bf f;
  ((float4*)&f)[0] = *(const float4*)(t + n*ld + h*8);
  ((float4*)&f)[1] = *(const float4*)(t + n*ld + 16 + h*8);
  return f;
}

// ---------------------------------------------------------------------------
// Generic bf16 GEMM:  C[M][N] = A[M][K] * Bt[N][K]^T  (+bias +resid, relu)
// Block 128x128, K-step 32, 256 threads = 8 waves (2x4), wave tile 64x32.
// Double-buffered LDS staging via async global->LDS copies.
// ---------------------------------------------------------------------------
template<int FLAGS>
__global__ void __launch_bounds__(256)
gemm_kernel(const bhalf* __restrict__ A, const bhalf* __restrict__ Bt,
            const float* __restrict__ bias, const float* __restrict__ resid,
            float* __restrict__ Cf, bhalf* __restrict__ Cb,
            int Ksz, int Nsz)
{
  __shared__ __align__(16) bhalf As[2][128*32];
  __shared__ __align__(16) bhalf Bs[2][128*32];
  const int tid = threadIdx.x;
  const int wid = tid >> 5;
  const int wm  = (wid >> 2) * 64;
  const int wn  = (wid & 3)  * 32;
  const int m0  = blockIdx.y * 128;
  const int n0  = blockIdx.x * 128;

  v8f acc[4][2];
  #pragma unroll
  for (int mi=0; mi<4; ++mi)
    #pragma unroll
    for (int ni=0; ni<2; ++ni) acc[mi][ni] = {};

  auto stage = [&](int buf, int k0){
    #pragma unroll
    for (int i=0; i<2; ++i){
      const int c   = tid + i*256;          // 0..511
      const int row = c >> 2;               // 0..127
      const int off = (c & 3) * 8;          // 0,8,16,24
      copy16(&A [(size_t)(m0+row)*Ksz + k0 + off], &As[buf][row*32+off]);
      copy16(&Bt[(size_t)(n0+row)*Ksz + k0 + off], &Bs[buf][row*32+off]);
    }
  };

  const int nk = Ksz >> 5;
  stage(0, 0);
  async_wait0();
  __syncthreads();

  for (int kt = 0; kt < nk; ++kt){
    const int cur = kt & 1;
    if (kt + 1 < nk) stage(cur ^ 1, (kt+1) << 5);   // overlap with compute

    v16bf af[4], bfv[2];
    #pragma unroll
    for (int mi=0; mi<4; ++mi) af[mi]  = frag_a(&As[cur][(wm+mi*16)*32], 32);
    #pragma unroll
    for (int ni=0; ni<2; ++ni) bfv[ni] = frag_b(&Bs[cur][(wn+ni*16)*32], 32);
    #pragma unroll
    for (int mi=0; mi<4; ++mi)
      #pragma unroll
      for (int ni=0; ni<2; ++ni)
        acc[mi][ni] = wmma_bf16(af[mi], bfv[ni], acc[mi][ni]);

    async_wait0();
    __syncthreads();
  }

  // Epilogue. C/D layout: VGPR j -> row j + 8*(lane/16), col = lane&15.
  const int l = tid & 31, cn = l & 15, h = l >> 4;
  #pragma unroll
  for (int ni=0; ni<2; ++ni){
    const int col = n0 + wn + ni*16 + cn;
    const float bv = (FLAGS & F_BIAS) ? bias[col] : 0.f;
    #pragma unroll
    for (int mi=0; mi<4; ++mi){
      const int rb = m0 + wm + mi*16 + h*8;
      #pragma unroll
      for (int j=0; j<8; ++j){
        const int row = rb + j;
        float v = acc[mi][ni][j] + bv;
        if (FLAGS & F_RELU)  v = fmaxf(v, 0.f);
        if (FLAGS & F_RESID) v += resid[(size_t)row*Nsz + col];
        if (FLAGS & F_BF16)  Cb[(size_t)row*Nsz + col] = (bhalf)v;
        else                 Cf[(size_t)row*Nsz + col] = v;
      }
    }
  }
}

// ---------------------------------------------------------------------------
// conv1 as implicit GEMM over 96 pipelined K-steps (3 taps x 32 D-tiles):
// h[m][f] = relu(b1[f] + sum_kk sum_d x[m+kk-1][d] * w1T[kk][f][d])
// ---------------------------------------------------------------------------
__global__ void __launch_bounds__(256)
conv1_kernel(const bhalf* __restrict__ xb, const bhalf* __restrict__ w1T,
             const float* __restrict__ b1, bhalf* __restrict__ hb)
{
  __shared__ __align__(16) bhalf As[2][128*32];
  __shared__ __align__(16) bhalf Bs[2][128*32];
  const int tid = threadIdx.x;
  const int wid = tid >> 5;
  const int wm  = (wid >> 2) * 64;
  const int wn  = (wid & 3)  * 32;
  const int m0  = blockIdx.y * 128;
  const int n0  = blockIdx.x * 128;

  v8f acc[4][2];
  #pragma unroll
  for (int mi=0; mi<4; ++mi)
    #pragma unroll
    for (int ni=0; ni<2; ++ni) acc[mi][ni] = {};

  auto stage = [&](int buf, int st){
    const int kk = st >> 5;            // tap 0..2
    const int k0 = (st & 31) << 5;     // D offset
    const bhalf* Bt = w1T + (size_t)kk * FFD * DD;
    #pragma unroll
    for (int i=0; i<2; ++i){
      const int c   = tid + i*256;
      const int row = c >> 2;
      const int off = (c & 3) * 8;
      const int m   = m0 + row;
      const int s2  = (m & (SS-1)) + kk - 1;
      bhalf* dstA = &As[buf][row*32+off];
      if ((unsigned)s2 < (unsigned)SS)
        copy16(&xb[(size_t)(m + kk - 1)*DD + k0 + off], dstA);
      else {
        float4 z = make_float4(0.f,0.f,0.f,0.f);
        *(float4*)dstA = z;            // bf16 zeros bitwise
      }
      copy16(&Bt[(size_t)(n0+row)*DD + k0 + off], &Bs[buf][row*32+off]);
    }
  };

  const int NSTEP = 3 * (DD/32);       // 96
  stage(0, 0);
  async_wait0();
  __syncthreads();

  for (int st=0; st<NSTEP; ++st){
    const int cur = st & 1;
    if (st + 1 < NSTEP) stage(cur ^ 1, st + 1);

    v16bf af[4], bfv[2];
    #pragma unroll
    for (int mi=0; mi<4; ++mi) af[mi]  = frag_a(&As[cur][(wm+mi*16)*32], 32);
    #pragma unroll
    for (int ni=0; ni<2; ++ni) bfv[ni] = frag_b(&Bs[cur][(wn+ni*16)*32], 32);
    #pragma unroll
    for (int mi=0; mi<4; ++mi)
      #pragma unroll
      for (int ni=0; ni<2; ++ni)
        acc[mi][ni] = wmma_bf16(af[mi], bfv[ni], acc[mi][ni]);

    async_wait0();
    __syncthreads();
  }

  const int l = tid & 31, cn = l & 15, h = l >> 4;
  #pragma unroll
  for (int ni=0; ni<2; ++ni){
    const int col = n0 + wn + ni*16 + cn;
    const float bv = b1[col];
    #pragma unroll
    for (int mi=0; mi<4; ++mi){
      const int rb = m0 + wm + mi*16 + h*8;
      #pragma unroll
      for (int j=0; j<8; ++j){
        float v = fmaxf(acc[mi][ni][j] + bv, 0.f);
        hb[(size_t)(rb+j)*FFD + col] = (bhalf)v;
      }
    }
  }
}

// ---------------------------------------------------------------------------
// Flash attention. Grid (S/64, B*H), 128 threads = 4 waves; each wave owns a
// 16-row q tile, streams 32-key blocks. Scores and P*V both via bf16 WMMA.
// ---------------------------------------------------------------------------
__global__ void __launch_bounds__(128)
attn_kernel(const bhalf* __restrict__ q, const bhalf* __restrict__ k,
            const bhalf* __restrict__ v, const float* __restrict__ mask,
            bhalf* __restrict__ ctx)
{
  __shared__ __align__(16) bhalf Qs[64*64];
  __shared__ __align__(16) bhalf Ks[32*64];      // [key][dh]  (== Bt for scores)
  __shared__ __align__(16) bhalf Vs[64*32];      // [dh][key]  (== Bt for P*V)
  __shared__ __align__(16) bhalf Ps[4][16*32];   // per-wave P tile

  const int tid = threadIdx.x, w = tid >> 5, l = tid & 31;
  const int cn = l & 15, h2 = l >> 4;
  const int hh = blockIdx.y & (HHN-1);
  const int b  = blockIdx.y >> 4;
  const int q0 = blockIdx.x * 64;
  const size_t rowbase = (size_t)b * SS;
  const bhalf* qg = q + rowbase*1024 + hh*64;
  const bhalf* kg = k + rowbase*1024 + hh*64;
  const bhalf* vg = v + rowbase*1024 + hh*64;
  const float* mg = mask + rowbase;

  // stage the 64x64 Q block once (async path)
  #pragma unroll
  for (int i=0; i<4; ++i){
    const int c = tid + i*128, r = c >> 3, off = (c & 7) * 8;
    copy16(&qg[(size_t)(q0+r)*1024 + off], &Qs[r*64+off]);
  }
  async_wait0();
  __syncthreads();
  const v16bf qa0 = frag_a(&Qs[(w*16)*64],      64);   // dh 0..31
  const v16bf qa1 = frag_a(&Qs[(w*16)*64 + 32], 64);   // dh 32..63

  v8f ctxa[4];
  #pragma unroll
  for (int t=0; t<4; ++t) ctxa[t] = {};
  float mrun[8], lrun[8];
  #pragma unroll
  for (int j=0; j<8; ++j){ mrun[j] = -1e30f; lrun[j] = 0.f; }
  const float scale = 0.125f;   // 1/sqrt(64)

  for (int it=0; it<SS/32; ++it){
    __syncthreads();
    const int kb = it*32;
    #pragma unroll
    for (int i=0; i<2; ++i){
      const int c = tid + i*128, r = c >> 3, off = (c & 7) * 8;
      copy16(&kg[(size_t)(kb+r)*1024 + off], &Ks[r*64+off]);
      float4 t4 = *(const float4*)&vg[(size_t)(kb+r)*1024 + off];
      const bhalf* e = (const bhalf*)&t4;
      #pragma unroll
      for (int u=0; u<8; ++u) Vs[(off+u)*32 + r] = e[u];   // transpose into [dh][key]
    }
    async_wait0();
    __syncthreads();

    v8f sc0 = {}, sc1 = {};
    sc0 = wmma_bf16(qa0, frag_b(&Ks[0*64],       64), sc0);
    sc0 = wmma_bf16(qa1, frag_b(&Ks[0*64 + 32],  64), sc0);
    sc1 = wmma_bf16(qa0, frag_b(&Ks[16*64],      64), sc1);
    sc1 = wmma_bf16(qa1, frag_b(&Ks[16*64 + 32], 64), sc1);

    const float mk0 = mg[kb + cn];
    const float mk1 = mg[kb + 16 + cn];
    float alpha[8];
    #pragma unroll
    for (int j=0; j<8; ++j){
      float a0 = sc0[j]*scale + mk0;
      float a1 = sc1[j]*scale + mk1;
      float mx = fmaxf(a0, a1);
      mx = fmaxf(mx, __shfl_xor(mx, 1, 32));
      mx = fmaxf(mx, __shfl_xor(mx, 2, 32));
      mx = fmaxf(mx, __shfl_xor(mx, 4, 32));
      mx = fmaxf(mx, __shfl_xor(mx, 8, 32));
      const float mn = fmaxf(mrun[j], mx);
      const float al = __expf(mrun[j] - mn);
      mrun[j] = mn;
      const float p0 = __expf(a0 - mn);
      const float p1 = __expf(a1 - mn);
      float rs = p0 + p1;
      rs += __shfl_xor(rs, 1, 32);
      rs += __shfl_xor(rs, 2, 32);
      rs += __shfl_xor(rs, 4, 32);
      rs += __shfl_xor(rs, 8, 32);
      lrun[j] = lrun[j]*al + rs;
      alpha[j] = al;
      Ps[w][(j + 8*h2)*32 + cn]      = (bhalf)p0;
      Ps[w][(j + 8*h2)*32 + 16 + cn] = (bhalf)p1;
    }
    #pragma unroll
    for (int t=0; t<4; ++t)
      #pragma unroll
      for (int j=0; j<8; ++j) ctxa[t][j] *= alpha[j];

    asm volatile("" ::: "memory");   // keep P stores before the fragment reload
    const v16bf pa = frag_a(&Ps[w][0], 32);
    #pragma unroll
    for (int t=0; t<4; ++t)
      ctxa[t] = wmma_bf16(pa, frag_b(&Vs[(t*16)*32], 32), ctxa[t]);
  }

  bhalf* og = ctx + rowbase*1024 + hh*64;
  #pragma unroll
  for (int j=0; j<8; ++j){
    const float inv = 1.0f / lrun[j];
    const int row = q0 + w*16 + j + 8*h2;
    #pragma unroll
    for (int t=0; t<4; ++t)
      og[(size_t)row*1024 + t*16 + cn] = (bhalf)(ctxa[t][j]*inv);
  }
}

// ---------------------------------------------------------------------------
// LayerNorm over D=1024 (residual already added into y). Writes fp32 + bf16.
// ---------------------------------------------------------------------------
__global__ void __launch_bounds__(256)
ln_kernel(const float* __restrict__ y, const float* __restrict__ g,
          const float* __restrict__ bta, float* __restrict__ xo,
          bhalf* __restrict__ xob)
{
  const int row = blockIdx.x, tid = threadIdx.x;
  const float* yr = y + (size_t)row*DD;
  float v[4], s = 0.f, s2 = 0.f;
  #pragma unroll
  for (int i=0; i<4; ++i){ v[i] = yr[tid + i*256]; s += v[i]; s2 += v[i]*v[i]; }
  #pragma unroll
  for (int m=1; m<32; m<<=1){ s += __shfl_xor(s, m, 32); s2 += __shfl_xor(s2, m, 32); }
  __shared__ float rs_[8], rs2_[8];
  const int wid = tid >> 5;
  if ((tid & 31) == 0){ rs_[wid] = s; rs2_[wid] = s2; }
  __syncthreads();
  float ts = 0.f, ts2 = 0.f;
  #pragma unroll
  for (int i=0; i<8; ++i){ ts += rs_[i]; ts2 += rs2_[i]; }
  const float mu  = ts * (1.f/DD);
  const float var = ts2 * (1.f/DD) - mu*mu;
  const float rsv = rsqrtf(var + 1e-5f);
  #pragma unroll
  for (int i=0; i<4; ++i){
    const int c = tid + i*256;
    const float o = (v[i]-mu)*rsv*g[c] + bta[c];
    xo [(size_t)row*DD + c] = o;
    xob[(size_t)row*DD + c] = (bhalf)o;
  }
}

// ---------------------------------------------------------------------------
// Prep kernels
// ---------------------------------------------------------------------------
__global__ void initx_kernel(const float* __restrict__ x, float* __restrict__ xo,
                             bhalf* __restrict__ xb, int n){
  const int i = blockIdx.x*256 + threadIdx.x;
  if (i < n){ const float t = x[i]; xo[i] = t; xb[i] = (bhalf)t; }
}
__global__ void cvt_kernel(const float* __restrict__ s, bhalf* __restrict__ d, int n){
  const int i = blockIdx.x*256 + threadIdx.x;
  if (i < n) d[i] = (bhalf)s[i];
}
// transpose+convert: dst[c][r] = src[r][c]; R,C multiples of 32
__global__ void tcvt_kernel(const float* __restrict__ src, bhalf* __restrict__ dst,
                            int R, int C){
  __shared__ float tile[32][33];
  const int tx = threadIdx.x & 31, ty = threadIdx.x >> 5;
  const int r0 = blockIdx.y*32, c0 = blockIdx.x*32;
  #pragma unroll
  for (int i=0; i<4; ++i){
    const int r = ty + i*8;
    tile[r][tx] = src[(size_t)(r0+r)*C + c0 + tx];
  }
  __syncthreads();
  #pragma unroll
  for (int i=0; i<4; ++i){
    const int r = ty + i*8;
    dst[(size_t)(c0+r)*R + r0 + tx] = (bhalf)tile[tx][r];
  }
}
// w1 (FF,D,3) -> w1T[kk][ff][d]
__global__ void w1cvt_kernel(const float* __restrict__ src, bhalf* __restrict__ dst){
  const int i = blockIdx.x*256 + threadIdx.x;       // over FF*D
  if (i >= FFD*DD) return;
  #pragma unroll
  for (int kk=0; kk<3; ++kk)
    dst[(size_t)kk*FFD*DD + i] = (bhalf)src[(size_t)i*3 + kk];
}

// ---------------------------------------------------------------------------
// Host orchestration
// ---------------------------------------------------------------------------
extern "C" void kernel_launch(void* const* d_in, const int* in_sizes, int n_in,
                              void* d_out, int out_size, void* d_ws, size_t ws_size,
                              hipStream_t stream)
{
  const float* x     = (const float*)d_in[0];
  const float* amask = (const float*)d_in[1];
  const float* Wq    = (const float*)d_in[2];
  const float* bq    = (const float*)d_in[3];
  const float* Wk    = (const float*)d_in[4];
  const float* bk    = (const float*)d_in[5];
  const float* Wv    = (const float*)d_in[6];
  const float* bv    = (const float*)d_in[7];
  const float* Wo    = (const float*)d_in[8];
  const float* bo    = (const float*)d_in[9];
  const float* ln1g  = (const float*)d_in[10];
  const float* ln1b  = (const float*)d_in[11];
  const float* w1    = (const float*)d_in[12];
  const float* b1    = (const float*)d_in[13];
  const float* w2    = (const float*)d_in[14];
  const float* b2    = (const float*)d_in[15];
  const float* ln2g  = (const float*)d_in[16];
  const float* ln2b  = (const float*)d_in[17];

  char* p = (char*)d_ws;
  auto alloc = [&](size_t bytes)->char*{
    char* r = p; p += (bytes + 255) & ~(size_t)255; return r;
  };
  const size_t MD = (size_t)MTOK * DD;
  float* x_f32 = (float*)alloc(MD*4);
  float* y_f32 = (float*)alloc(MD*4);
  bhalf* xb    = (bhalf*)alloc(MD*2);
  bhalf* qb    = (bhalf*)alloc(MD*2);
  bhalf* kb2   = (bhalf*)alloc(MD*2);
  bhalf* vb2   = (bhalf*)alloc(MD*2);
  bhalf* ctxb  = (bhalf*)alloc(MD*2);
  bhalf* hb    = (bhalf*)alloc((size_t)MTOK*FFD*2);
  bhalf* wqT   = (bhalf*)alloc((size_t)DD*DD*2);
  bhalf* wkT   = (bhalf*)alloc((size_t)DD*DD*2);
  bhalf* wvT   = (bhalf*)alloc((size_t)DD*DD*2);
  bhalf* woT   = (bhalf*)alloc((size_t)DD*DD*2);
  bhalf* w1T   = (bhalf*)alloc((size_t)3*FFD*DD*2);
  bhalf* w2b   = (bhalf*)alloc((size_t)DD*FFD*2);

  initx_kernel<<<(int)(MD/256), 256, 0, stream>>>(x, x_f32, xb, (int)MD);

  const dim3 gq(DD/128, MTOK/128);     // (8, 64)   N=1024 GEMMs
  const dim3 gc1(FFD/128, MTOK/128);   // (32, 64)  conv1
  const dim3 ga(SS/64, BB*HHN);        // (32, 64)  attention
  const dim3 tg(DD/32, DD/32);         // (32, 32)  1024x1024 transposes

  for (int ll=0; ll<LLN; ++ll){
    // weight prep for this layer (bf16, B-tile friendly [N][K] layouts)
    tcvt_kernel<<<tg, 256, 0, stream>>>(Wq + (size_t)ll*DD*DD, wqT, DD, DD);
    tcvt_kernel<<<tg, 256, 0, stream>>>(Wk + (size_t)ll*DD*DD, wkT, DD, DD);
    tcvt_kernel<<<tg, 256, 0, stream>>>(Wv + (size_t)ll*DD*DD, wvT, DD, DD);
    tcvt_kernel<<<tg, 256, 0, stream>>>(Wo + (size_t)ll*DD*DD, woT, DD, DD);
    w1cvt_kernel<<<(FFD*DD)/256, 256, 0, stream>>>(w1 + (size_t)ll*FFD*DD*3, w1T);
    cvt_kernel<<<(DD*FFD)/256, 256, 0, stream>>>(w2 + (size_t)ll*DD*FFD, w2b, DD*FFD);

    // QKV projections
    gemm_kernel<F_BIAS|F_BF16><<<gq, 256, 0, stream>>>(
        xb, wqT, bq + (size_t)ll*DD, nullptr, nullptr, qb, DD, DD);
    gemm_kernel<F_BIAS|F_BF16><<<gq, 256, 0, stream>>>(
        xb, wkT, bk + (size_t)ll*DD, nullptr, nullptr, kb2, DD, DD);
    gemm_kernel<F_BIAS|F_BF16><<<gq, 256, 0, stream>>>(
        xb, wvT, bv + (size_t)ll*DD, nullptr, nullptr, vb2, DD, DD);

    // attention
    attn_kernel<<<ga, 128, 0, stream>>>(qb, kb2, vb2, amask, ctxb);

    // output projection + residual
    gemm_kernel<F_BIAS|F_RESID><<<gq, 256, 0, stream>>>(
        ctxb, woT, bo + (size_t)ll*DD, x_f32, y_f32, nullptr, DD, DD);
    ln_kernel<<<MTOK, 256, 0, stream>>>(y_f32, ln1g + (size_t)ll*DD,
                                        ln1b + (size_t)ll*DD, x_f32, xb);

    // conv FFN
    conv1_kernel<<<gc1, 256, 0, stream>>>(xb, w1T, b1 + (size_t)ll*FFD, hb);
    gemm_kernel<F_BIAS|F_RESID><<<gq, 256, 0, stream>>>(
        hb, w2b, b2 + (size_t)ll*DD, x_f32, y_f32, nullptr, FFD, DD);
    float* lndst = (ll == LLN-1) ? (float*)d_out : x_f32;
    ln_kernel<<<MTOK, 256, 0, stream>>>(y_f32, ln2g + (size_t)ll*DD,
                                        ln2b + (size_t)ll*DD, lndst, xb);
  }
}
